// FixedDiagonalLDS_10917806867217
// MI455X (gfx1250) — matrix operand order repaired and checked by reading, not compiled, not run on hardware
//
#include <hip/hip_runtime.h>
#include <hip/hip_bf16.h>

// Problem dims (fixed by the reference)
#define NB 16       // batch
#define TT 4096     // time
#define DD 512      // state dim (GEMM K)
#define OO 512      // output dim (GEMM N)
#define MM (NB*TT)  // GEMM M = 65536

typedef __attribute__((ext_vector_type(16))) _Float16 v16h;
typedef __attribute__((ext_vector_type(8)))  float    v8f;
typedef __attribute__((address_space(3)))    _Float16 lds_h;

// gfx1250 async global->LDS copy (ASYNCcnt-tracked), 16 bytes per lane.
__device__ __forceinline__ void async_b128(const _Float16* g, lds_h* l) {
  asm volatile("global_load_async_to_lds_b128 %0, %1, off"
               :: "v"(l), "v"(g) : "memory");
}

// ---------------------------------------------------------------------------
// Kernel 1: diagonal scan h_{t+1} = A*h_t + B*u_t, emitting h directly in
// WMMA A-fragment order (16x32 f16 A tile, ISA 7.12.2 layout).
// One thread per (n,d); t-loop is sequential (the only true dependency).
// ---------------------------------------------------------------------------
__global__ __launch_bounds__(256) void lds_scan_kernel(
    const float* __restrict__ u, const float* __restrict__ A,
    const float* __restrict__ Bv, const float* __restrict__ h0,
    _Float16* __restrict__ Hfrag)
{
  int tid = blockIdx.x * blockDim.x + threadIdx.x;   // n*DD + d
  int d = tid & (DD - 1);
  int n = tid >> 9;                                  // DD = 512
  float a = A[d];
  float b = Bv[d];
  float h = h0[d];
  int tk = d >> 5;        // K-tile (K step 32)
  int kk = d & 31;        // k within tile
  int g, i;
  if      (kk < 8)  { g = 0; i = kk;      }
  else if (kk < 16) { g = 1; i = kk - 8;  }
  else if (kk < 24) { g = 0; i = kk - 8;  }
  else              { g = 1; i = kk - 16; }
  const float* urow = u + (long)n * TT;
  int laneBase = (g << 8) + i;                       // + m*16 per step
  for (int t = 0; t < TT; ++t) {
    h = a * h + b * urow[t];
    int m  = t & 15;
    int tm = (n << 8) + (t >> 4);                    // row tile of r = n*TT+t
    int off = ((tm << 4) + tk) * 512 + laneBase + (m << 4);
    Hfrag[off] = (_Float16)h;
  }
}

// ---------------------------------------------------------------------------
// Kernel 2: convert C (D x O, f32) into WMMA B-fragment order (32x16 f16 B):
// fragment (tk,tn) at ((tk*32+tn)*512); lane l holds column nn=l%16,
// K = i<8 ? (l/16)*8 + i : 16 + (l/16)*8 + (i-8)  -> table K order per lane.
// ---------------------------------------------------------------------------
__global__ __launch_bounds__(256) void convC_kernel(
    const float* __restrict__ C, _Float16* __restrict__ Cfrag)
{
  int idx = blockIdx.x * blockDim.x + threadIdx.x;   // d*OO + o
  int d = idx >> 9;
  int o = idx & (OO - 1);
  int tk = d >> 5, dd = d & 31, g = (dd >> 3) & 1, i = (dd & 7) | ((dd >> 4) << 3);
  int tn = o >> 4, nn = o & 15;
  int l  = (g << 4) + nn;
  int off = ((tk << 5) + tn) * 512 + l * 16 + i;
  Cfrag[off] = (_Float16)C[idx];
}

// ---------------------------------------------------------------------------
// Kernel 3: GEMM  out[r,o] = sum_d Hf16[r,d] * Cf16[d,o], f32 accumulate.
// Block = 256 threads = 8 wave32s; block tile 64(M) x 256(N).
// Wave (wm 0..1, wn 0..3): rows [wm*32,+32), cols [wn*64,+64)
//   -> 2 A frags x 4 B frags = 8 WMMAs per k-step per wave.
// B fragments staged to double-buffered LDS with GLOBAL_LOAD_ASYNC_TO_LDS
// (ASYNCcnt), overlapped with the WMMA work of the previous k-step.
// ---------------------------------------------------------------------------
__global__ __launch_bounds__(256) void gemm_wmma_kernel(
    const _Float16* __restrict__ Hfrag, const _Float16* __restrict__ Cfrag,
    float* __restrict__ out)
{
  __shared__ __align__(32) _Float16 ldsB[2][16 * 512];   // 2 x 16 KB

  int lane = threadIdx.x & 31;
  int w    = threadIdx.x >> 5;
  int wm   = w & 1;        // 0..1 : 32 rows each
  int wn   = w >> 1;       // 0..3 : 64 cols each
  int tmG0   = blockIdx.x * 4 + wm * 2;   // first of the wave's 2 M-tiles
  int tnBase = blockIdx.y * 16;           // block's first N-tile (of 32)
  int lf0    = 2 * w;                     // wave stages local frags 2w, 2w+1

  v8f acc[2][4] = {};

  // prologue: stage k-step 0 into buffer 0 (4 x b128 per lane = 16 KB/block)
  {
    const _Float16* src = Cfrag + ((tnBase + lf0) << 9) + lane * 16;
    lds_h* dst = (lds_h*)&ldsB[0][(lf0 << 9) + lane * 16];
    async_b128(src,       dst);
    async_b128(src + 8,   dst + 8);
    async_b128(src + 512, dst + 512);
    async_b128(src + 520, dst + 520);
  }

  for (int kt = 0; kt < 16; ++kt) {
    asm volatile("s_wait_asynccnt 0x0" ::: "memory");  // my stage-kt chunks landed
    __syncthreads();                                   // everyone's landed; other buf free
    if (kt < 15) {
      const _Float16* src =
          Cfrag + ((((kt + 1) << 5) + tnBase + lf0) << 9) + lane * 16;
      lds_h* dst = (lds_h*)&ldsB[(kt + 1) & 1][(lf0 << 9) + lane * 16];
      async_b128(src,       dst);
      async_b128(src + 8,   dst + 8);
      async_b128(src + 512, dst + 512);
      async_b128(src + 520, dst + 520);
      __builtin_prefetch(Hfrag + (((tmG0 << 4) + kt + 1) << 9), 0, 3);
    }

    const _Float16* abase = Hfrag + (((tmG0 << 4) + kt) << 9) + lane * 16;
    v16h a0 = *(const v16h*)abase;           // M-tile tmG0
    v16h a1 = *(const v16h*)(abase + 8192);  // M-tile tmG0+1 (16 frags later)

    const _Float16* bbase = &ldsB[kt & 1][((wn * 4) << 9) + lane * 16];
#pragma unroll
    for (int j = 0; j < 4; ++j) {
      v16h b = *(const v16h*)(bbase + (j << 9));
      acc[0][j] = __builtin_amdgcn_wmma_f32_16x16x32_f16(
          false, a0, false, b, (short)0, acc[0][j], false, false);
      acc[1][j] = __builtin_amdgcn_wmma_f32_16x16x32_f16(
          false, a1, false, b, (short)0, acc[1][j], false, false);
    }
  }

  // Store: C/D layout — lane l holds column nn=l%16; VGPR v holds row
  // M = v + 8*(l/16).  out[r*OO + o].
  int rowBase = blockIdx.x * 64 + wm * 32 + ((lane >> 4) << 3);
  int colBase = blockIdx.y * 256 + wn * 64 + (lane & 15);
#pragma unroll
  for (int mi = 0; mi < 2; ++mi) {
#pragma unroll
    for (int j = 0; j < 4; ++j) {
      long col = colBase + j * 16;
#pragma unroll
      for (int v = 0; v < 8; ++v) {
        out[(long)(rowBase + mi * 16 + v) * OO + col] = acc[mi][j][v];
      }
    }
  }
}

// ---------------------------------------------------------------------------
extern "C" void kernel_launch(void* const* d_in, const int* in_sizes, int n_in,
                              void* d_out, int out_size, void* d_ws, size_t ws_size,
                              hipStream_t stream) {
  const float* u  = (const float*)d_in[0];   // (NB, TT)
  const float* A  = (const float*)d_in[1];   // (DD,)
  const float* Bv = (const float*)d_in[2];   // (DD,)
  const float* C  = (const float*)d_in[3];   // (DD, OO)
  const float* h0 = (const float*)d_in[4];   // (DD,)
  float* out = (float*)d_out;                // (NB, TT, OO) f32

  _Float16* Hfrag = (_Float16*)d_ws;
  _Float16* Cfrag = (_Float16*)((char*)d_ws + (size_t)MM * DD * sizeof(_Float16));

  lds_scan_kernel<<<(NB * DD) / 256, 256, 0, stream>>>(u, A, Bv, h0, Hfrag);
  convC_kernel<<<(DD * OO) / 256, 256, 0, stream>>>(C, Cfrag);

  dim3 grid(MM / 64, OO / 256);
  gemm_wmma_kernel<<<grid, 256, 0, stream>>>(Hfrag, Cfrag, out);
}
